// StripAttentionModule_43860206027366
// MI455X (gfx1250) — matrix operand order repaired and checked by reading, not compiled, not run on hardware
//
#include <hip/hip_runtime.h>

typedef __attribute__((ext_vector_type(8)))  float  v8f;
typedef __attribute__((ext_vector_type(8)))  __bf16 v8bf;
typedef __attribute__((ext_vector_type(16))) __bf16 v16bf;

// ---------------------------------------------------------------------------
// WMMA helpers (CDNA5 gfx1250, wave32). Layouts used here (A/B K order is this code's own):
//  A (16x32 bf16):  lane m=lane%16; 16 contiguous K at kt*32 + hi*16.
//  B (32x16 bf16):  lane n=lane%16; 16 contiguous K at kt*32 + hi*16.
//  C/D (16x16 f32): lane holds M = j + 8*hi (j = vgpr), N = lane%16.
// ---------------------------------------------------------------------------
__device__ __forceinline__ v16bf afrag(const __bf16* base, int m, int ldk,
                                       int kt, int hi) {
  return *(const v16bf*)(base + m * ldk + kt * 32 + hi * 16);
}

__device__ __forceinline__ v16bf bfrag(const __bf16* colmaj, int n, int ldk,
                                       int kt, int hi) {
  return *(const v16bf*)(colmaj + n * ldk + kt * 32 + hi * 16);
}

__device__ __forceinline__ v8f wmma_bf16(v16bf a, v16bf b, v8f c) {
  return __builtin_amdgcn_wmma_f32_16x16x32_bf16(
      /*neg_a=*/false, a, /*neg_b=*/false, b,
      /*c_mod=*/(short)0, c, /*reuse_a=*/false, /*reuse_b=*/false);
}

// ---------------------------------------------------------------------------
// Pass 1: q32[b][64][128] = mean_h relu(sq*(Wq x)+bq),
//         v32[b][128][128] = mean_h relu(sv*(Wv x)+bv)   (atomic f32 partials)
// grid (wtile=8, b=16, hchunk=4), 128 threads = 4 waves.
// Each wave owns 3 of the 12 output M-tiles (4 for q, 8 for v).
// ---------------------------------------------------------------------------
__global__ __launch_bounds__(128) void strip_qv_kernel(
    const float* __restrict__ x, const float* __restrict__ Wq,
    const float* __restrict__ sq, const float* __restrict__ bq,
    const float* __restrict__ Wv, const float* __restrict__ sv,
    const float* __restrict__ bv, float* __restrict__ q32,
    float* __restrict__ v32) {
  extern __shared__ __bf16 smem[];
  __bf16* Wq_l = smem;            // 64*128
  __bf16* Wv_l = Wq_l + 64 * 128; // 128*128
  __bf16* X_l  = Wv_l + 128 * 128; // 16*128  [w][c], c contiguous

  const int tid = threadIdx.x;
  const int b   = blockIdx.y;
  const int w0  = blockIdx.x * 16;
  const int h0  = blockIdx.z * 32;

  for (int i = tid; i < 64 * 128; i += 128)  Wq_l[i] = (__bf16)Wq[i];
  for (int i = tid; i < 128 * 128; i += 128) Wv_l[i] = (__bf16)Wv[i];
  __syncthreads();

  const int wv = tid >> 5, lane = tid & 31, lo = lane & 15, hi = lane >> 4;

  v8f acc[3] = {};
  v8f sreg[3], breg[3];
  const __bf16* Abase[3];
  int mbase[3];
  bool isq[3];
#pragma unroll
  for (int t = 0; t < 3; ++t) {
    int g = wv + 4 * t;
    if (g < 4) {
      isq[t] = true;  mbase[t] = g * 16;
      Abase[t] = Wq_l + g * 16 * 128;
      sreg[t] = *(const v8f*)(sq + g * 16 + hi * 8);
      breg[t] = *(const v8f*)(bq + g * 16 + hi * 8);
    } else {
      isq[t] = false; mbase[t] = (g - 4) * 16;
      Abase[t] = Wv_l + (g - 4) * 16 * 128;
      sreg[t] = *(const v8f*)(sv + (g - 4) * 16 + hi * 8);
      breg[t] = *(const v8f*)(bv + (g - 4) * 16 + hi * 8);
    }
  }

  for (int h = h0; h < h0 + 32; ++h) {
    // stage X tile: thread tid = channel c, 16 contiguous w's, bf16 into [w][c]
    {
      const float* xp = x + (((size_t)b * 128 + tid) * 128 + h) * 128 + w0;
#pragma unroll
      for (int j = 0; j < 16; j += 4) {
        float4 vx = *(const float4*)(xp + j);
        X_l[(j + 0) * 128 + tid] = (__bf16)vx.x;
        X_l[(j + 1) * 128 + tid] = (__bf16)vx.y;
        X_l[(j + 2) * 128 + tid] = (__bf16)vx.z;
        X_l[(j + 3) * 128 + tid] = (__bf16)vx.w;
      }
    }
    __syncthreads();
#pragma unroll
    for (int t = 0; t < 3; ++t) {
      v8f y = {};
#pragma unroll
      for (int kt = 0; kt < 4; ++kt)
        y = wmma_bf16(afrag(Abase[t], lo, 128, kt, hi),
                      bfrag(X_l, lo, 128, kt, hi), y);
#pragma unroll
      for (int j = 0; j < 8; ++j) {
        float r = y[j] * sreg[t][j] + breg[t][j];
        acc[t][j] += (r > 0.f) ? r : 0.f;
      }
    }
    __syncthreads();
  }

#pragma unroll
  for (int t = 0; t < 3; ++t) {
#pragma unroll
    for (int j = 0; j < 8; ++j) {
      int m = mbase[t] + hi * 8 + j;
      float val = acc[t][j] * (1.0f / 128.0f);
      if (isq[t]) atomicAdd(q32 + ((size_t)b * 64 + m) * 128 + w0 + lo, val);
      else        atomicAdd(v32 + ((size_t)b * 128 + m) * 128 + w0 + lo, val);
    }
  }
}

// ---------------------------------------------------------------------------
// Pass 1.5: f32 -> bf16 in WMMA A-operand layouts: qT[b][w][c], v[b][c][w]
// ---------------------------------------------------------------------------
__global__ __launch_bounds__(256) void strip_cvt_kernel(
    const float* __restrict__ q32, const float* __restrict__ v32,
    __bf16* __restrict__ qT, __bf16* __restrict__ vb) {
  int i = blockIdx.x * 256 + threadIdx.x;
  const int nq = 16 * 64 * 128, nv = 16 * 128 * 128;
  if (i < nq) {
    int b = i >> 13, c = (i >> 7) & 63, w = i & 127;
    qT[((size_t)b * 128 + w) * 64 + c] = (__bf16)q32[i];
  }
  if (i < nv) vb[i] = (__bf16)v32[i];
}

// ---------------------------------------------------------------------------
// Pass 2: per wave: one 16-column n-tile.
//   k = relu(sk*(Wk x)+bk)       (4x4 WMMA,  D -> LDS [n][cmid])
//   s = qT k                     (8x2 WMMA)
//   softmax over w (in-register + shfl_xor(16))  -> p -> LDS [n][w]
//   aug = v p; out = x + aug     (8x4 WMMA)
// grid (256, 16), 128 threads = 4 waves.
// ---------------------------------------------------------------------------
__global__ __launch_bounds__(128) void strip_attn_kernel(
    const float* __restrict__ x, const float* __restrict__ Wk,
    const float* __restrict__ sk, const float* __restrict__ bk,
    const __bf16* __restrict__ qT, const __bf16* __restrict__ vb,
    float* __restrict__ out) {
  extern __shared__ __bf16 smem[];
  __bf16* Wk_l = smem;              //  64*128  row-major [cmid][c]
  __bf16* q_l  = Wk_l + 64 * 128;   // 128*64   row-major [w][c]
  __bf16* v_l  = q_l + 128 * 64;    // 128*128  row-major [c][w]
  __bf16* X_l  = v_l + 128 * 128;   // 4 waves * 16*128  [n][c]
  __bf16* k_l  = X_l + 4 * 16 * 128; // 4 waves * 16*64  [n][cmid]
  __bf16* p_l  = k_l + 4 * 16 * 64;  // 4 waves * 16*128 [n][w]

  const int tid = threadIdx.x;
  const int b   = blockIdx.y;

  for (int i = tid; i < 64 * 128; i += 128) Wk_l[i] = (__bf16)Wk[i];
  {
    const int4* qs = (const int4*)(qT + (size_t)b * 128 * 64);
    int4* qd = (int4*)q_l;
    for (int i = tid; i < (128 * 64) / 8; i += 128) qd[i] = qs[i];
    const int4* vs = (const int4*)(vb + (size_t)b * 128 * 128);
    int4* vd = (int4*)v_l;
    for (int i = tid; i < (128 * 128) / 8; i += 128) vd[i] = vs[i];
  }
  __syncthreads();

  const int wv = tid >> 5, lane = tid & 31, lo = lane & 15, hi = lane >> 4;
  const int n0 = (blockIdx.x * 4 + wv) * 16;
  __bf16* Xw = X_l + wv * 2048;
  __bf16* kw = k_l + wv * 1024;
  __bf16* pw = p_l + wv * 2048;

  // stage per-wave X tile (bf16, [n][c]); lanes split even/odd channels
  {
    const float* xb = x + ((size_t)b * 128) * 16384 + n0 + lo;
#pragma unroll 4
    for (int cc = 0; cc < 64; ++cc) {
      int c = cc * 2 + hi;
      Xw[lo * 128 + c] = (__bf16)xb[(size_t)c * 16384];
    }
  }

  // k-conv (same-wave LDS producer/consumer: DS ops are in-order per wave)
#pragma unroll
  for (int mt = 0; mt < 4; ++mt) {
    v8f y = {};
#pragma unroll
    for (int kt = 0; kt < 4; ++kt)
      y = wmma_bf16(afrag(Wk_l + mt * 16 * 128, lo, 128, kt, hi),
                    bfrag(Xw, lo, 128, kt, hi), y);
    v8f s  = *(const v8f*)(sk + mt * 16 + hi * 8);
    v8f bb = *(const v8f*)(bk + mt * 16 + hi * 8);
    v8bf kb;
#pragma unroll
    for (int j = 0; j < 8; ++j) {
      float r = y[j] * s[j] + bb[j];
      kb[j] = (__bf16)((r > 0.f) ? r : 0.f);
    }
    *(v8bf*)(kw + lo * 64 + mt * 16 + hi * 8) = kb;  // D -> [n][cmid]
  }

  // scores: M = w (8 tiles), K = cmid (2 steps)
  v8f sc[8];
#pragma unroll
  for (int mt = 0; mt < 8; ++mt) {
    v8f y = {};
#pragma unroll
    for (int kt = 0; kt < 2; ++kt)
      y = wmma_bf16(afrag(q_l + mt * 16 * 64, lo, 64, kt, hi),
                    bfrag(kw, lo, 64, kt, hi), y);
    sc[mt] = y;
  }

  // softmax over the 128 w's of each column n (lane pair L / L^16)
  float mx = -3.4e38f;
#pragma unroll
  for (int mt = 0; mt < 8; ++mt)
#pragma unroll
    for (int j = 0; j < 8; ++j) mx = fmaxf(mx, sc[mt][j]);
  mx = fmaxf(mx, __shfl_xor(mx, 16, 32));
  float sum = 0.f;
#pragma unroll
  for (int mt = 0; mt < 8; ++mt)
#pragma unroll
    for (int j = 0; j < 8; ++j) {
      float e = __expf(sc[mt][j] - mx);
      sc[mt][j] = e;
      sum += e;
    }
  sum += __shfl_xor(sum, 16, 32);
  float inv = 1.f / sum;
#pragma unroll
  for (int mt = 0; mt < 8; ++mt) {
    v8bf pb;
#pragma unroll
    for (int j = 0; j < 8; ++j) pb[j] = (__bf16)(sc[mt][j] * inv);
    *(v8bf*)(pw + lo * 128 + mt * 16 + hi * 8) = pb;  // p -> [n][w]
  }

  // aug = v p, residual add in fp32
#pragma unroll
  for (int mt = 0; mt < 8; ++mt) {
    v8f y = {};
#pragma unroll
    for (int kt = 0; kt < 4; ++kt)
      y = wmma_bf16(afrag(v_l + mt * 16 * 128, lo, 128, kt, hi),
                    bfrag(pw, lo, 128, kt, hi), y);
#pragma unroll
    for (int j = 0; j < 8; ++j) {
      int c = mt * 16 + hi * 8 + j;
      size_t idx = ((size_t)b * 128 + c) * 16384 + n0 + lo;
      out[idx] = x[idx] + y[j];
    }
  }
}

// ---------------------------------------------------------------------------
extern "C" void kernel_launch(void* const* d_in, const int* in_sizes, int n_in,
                              void* d_out, int out_size, void* d_ws,
                              size_t ws_size, hipStream_t stream) {
  const float* x  = (const float*)d_in[0];
  const float* Wq = (const float*)d_in[1];
  const float* sq = (const float*)d_in[2];
  const float* bq = (const float*)d_in[3];
  const float* Wk = (const float*)d_in[4];
  const float* sk = (const float*)d_in[5];
  const float* bk = (const float*)d_in[6];
  const float* Wv = (const float*)d_in[7];
  const float* sv = (const float*)d_in[8];
  const float* bv = (const float*)d_in[9];
  float* out = (float*)d_out;

  char* ws = (char*)d_ws;
  float*  q32 = (float*)ws;                    // 16*64*128  f32 (512 KB)
  float*  v32 = (float*)(ws + 524288);         // 16*128*128 f32 (1 MB)
  __bf16* qT  = (__bf16*)(ws + 1572864);       // 16*128*64  bf16
  __bf16* vb  = (__bf16*)(ws + 1835008);       // 16*128*128 bf16

  // atomic accumulators must be zeroed every call (graph replays don't reset ws)
  hipMemsetAsync(ws, 0, 1572864, stream);

  size_t sh1 = (size_t)(64 * 128 + 128 * 128 + 16 * 128) * sizeof(__bf16);
  hipFuncSetAttribute((const void*)strip_qv_kernel,
                      hipFuncAttributeMaxDynamicSharedMemorySize, (int)sh1);
  strip_qv_kernel<<<dim3(8, 16, 4), 128, sh1, stream>>>(x, Wq, sq, bq, Wv, sv,
                                                        bv, q32, v32);

  strip_cvt_kernel<<<dim3(1024), 256, 0, stream>>>(q32, v32, qT, vb);

  size_t sh2 = (size_t)(64 * 128 + 128 * 64 + 128 * 128 + 4 * 16 * 128 +
                        4 * 16 * 64 + 4 * 16 * 128) * sizeof(__bf16);
  hipFuncSetAttribute((const void*)strip_attn_kernel,
                      hipFuncAttributeMaxDynamicSharedMemorySize, (int)sh2);
  strip_attn_kernel<<<dim3(256, 16), 128, sh2, stream>>>(x, Wk, sk, bk, qT, vb,
                                                         out);
}
